// _MHA_23880018166195
// MI455X (gfx1250) — matrix operand order repaired and checked by reading, not compiled, not run on hardware
//
#include <hip/hip_runtime.h>

// ---------------------------------------------------------------------------
// Types for CDNA5 WMMA (wave32, 16x16x32 bf16 -> f32 accumulate) and TDM
// ---------------------------------------------------------------------------
typedef __attribute__((ext_vector_type(16))) __bf16 v16bf;
typedef __attribute__((ext_vector_type(8)))  __bf16 v8bf;
typedef __attribute__((ext_vector_type(8)))  float  v8f;
typedef __attribute__((ext_vector_type(4)))  unsigned int u32x4;
typedef __attribute__((ext_vector_type(8)))  int          i32x8;
typedef __attribute__((ext_vector_type(4)))  int          i32x4;

#if __has_builtin(__builtin_amdgcn_tensor_load_to_lds) && \
    __has_builtin(__builtin_amdgcn_s_wait_tensorcnt)
#define HAVE_TDM 1
#else
#define HAVE_TDM 0
#endif

__device__ inline v8f wmma_bf16(v16bf a, v16bf b, v8f c) {
    return __builtin_amdgcn_wmma_f32_16x16x32_bf16(
        /*neg_a=*/false, a, /*neg_b=*/false, b,
        /*c_mod=*/(short)0, c, /*reuse_a=*/false, /*reuse_b=*/false);
}

__device__ inline v16bf pack2(const __bf16* lo, const __bf16* hi) {
    v8bf a = *(const v8bf*)lo;
    v8bf b = *(const v8bf*)hi;
    v16bf r;
#pragma unroll
    for (int i = 0; i < 8; i++) { r[i] = a[i]; r[i + 8] = b[i]; }
    return r;
}

__device__ inline v8f zero8() {
    v8f z;
#pragma unroll
    for (int i = 0; i < 8; i++) z[i] = 0.0f;
    return z;
}

#if HAVE_TDM
// ---------------------------------------------------------------------------
// TDM: DMA one 2-D bf16 tile (tile_d1 rows x tile_d0 cols, row stride
// `stride0` elements) from global memory into LDS (row-major, packed).
// D# layout per CDNA5 ISA ch.8: group0 = {count/type/lds_addr/global_addr},
// group1 = {data_size, tensor dims, tile dims, dim0 stride}. 2-D tensor ->
// groups 2/3 (and trailing group) zero. This toolchain exposes the 6-arg
// builtin: (g0, g1, g2, g3, extra, cpol). Tracked by TENSORcnt.
// ---------------------------------------------------------------------------
__device__ inline void tdm_load_2d_bf16(unsigned int lds_off, const void* gptr,
                                        unsigned int tensor_d0,
                                        unsigned int tensor_d1,
                                        unsigned int tile_d0,
                                        unsigned int tile_d1,
                                        unsigned long long stride0_elems) {
    unsigned long long ga = (unsigned long long)(uintptr_t)gptr;
    u32x4 g0;
    g0[0] = 1u;                                   // count=1, user mode
    g0[1] = lds_off;                              // lds_addr [63:32]
    g0[2] = (unsigned int)ga;                     // global_addr low
    g0[3] = (unsigned int)((ga >> 32) & 0x01FFFFFFu) | (2u << 30); // type=2
    i32x8 g1;
    g1[0] = (int)(1u << 16);                      // data_size=1 (2 bytes)
    g1[1] = (int)((tensor_d0 & 0xFFFFu) << 16);   // tensor_dim0 [79:48] lo
    g1[2] = (int)((tensor_d0 >> 16) | ((tensor_d1 & 0xFFFFu) << 16));
    g1[3] = (int)((tensor_d1 >> 16) | (tile_d0 << 16));   // tile_dim0 [127:112]
    g1[4] = (int)(tile_d1 & 0xFFFFu);             // tile_dim1 [143:128]
    g1[5] = (int)(unsigned int)stride0_elems;     // tensor_dim0_stride lo32
    g1[6] = (int)(unsigned int)(stride0_elems >> 32);     // stride hi16
    g1[7] = 0;
    i32x4 z4 = {0, 0, 0, 0};
    i32x8 z8 = {0, 0, 0, 0, 0, 0, 0, 0};
    __builtin_amdgcn_tensor_load_to_lds(g0, g1, z4, z4, z8, 0);
}
#endif

#define BATCH   4
#define S_LEN   1024
#define DMODEL  1024
#define NHEADS  16
#define HDIM    64
#define BS_ROWS (BATCH * S_LEN)   // 4096

// ---------------------------------------------------------------------------
// f32 -> bf16 conversion
// ---------------------------------------------------------------------------
__global__ void cvt_f32_bf16_kernel(const float* __restrict__ src,
                                    __bf16* __restrict__ dst, int n) {
    int i = blockIdx.x * blockDim.x + threadIdx.x;
    if (i < n) dst[i] = (__bf16)src[i];
}

// ---------------------------------------------------------------------------
// Tiled WMMA GEMM: C[M,N] = A[M,K] * B[K,N], row-major bf16 in, f32 acc.
// 128 threads = 4 waves; tile 64x64, K-step 32. Wave w owns rows 16w..16w+15.
// Tiles are DMA'd row-major into LDS by wave 0 via TDM; B fragments are
// gathered with strided DS reads (transpose on the read side).
// ---------------------------------------------------------------------------
template <bool OUT_BF16>
__global__ __launch_bounds__(128) void gemm_wmma_kernel(
        const __bf16* __restrict__ A, const __bf16* __restrict__ B,
        void* __restrict__ Cout, int M, int N, int K) {
    __shared__ __align__(16) __bf16 lds_a[64][32];
    __shared__ __align__(16) __bf16 lds_b[32][64];

    const int tid  = threadIdx.x;
    const int lane = tid & 31;
    const int w    = tid >> 5;
    const int hl   = lane >> 4;   // lane/16
    const int ln   = lane & 15;   // lane%16
    const int m0 = blockIdx.x * 64;
    const int n0 = blockIdx.y * 64;

    v8f acc[4];
#pragma unroll
    for (int i = 0; i < 4; i++) acc[i] = zero8();

#if HAVE_TDM
    const unsigned int lds_off_a = (unsigned int)(uintptr_t)&lds_a[0][0];
    const unsigned int lds_off_b = (unsigned int)(uintptr_t)&lds_b[0][0];
#endif

    for (int k0 = 0; k0 < K; k0 += 32) {
#if HAVE_TDM
        if (w == 0) {
            // A tile: 64 rows x 32 cols, row stride K
            tdm_load_2d_bf16(lds_off_a, A + (size_t)m0 * K + k0,
                             (unsigned)K, (unsigned)M, 32u, 64u,
                             (unsigned long long)K);
            // B tile: 32 rows x 64 cols, row stride N
            tdm_load_2d_bf16(lds_off_b, B + (size_t)k0 * N + n0,
                             (unsigned)N, (unsigned)K, 64u, 32u,
                             (unsigned long long)N);
        }
        if (k0 + 32 < K)
            __builtin_prefetch(B + (size_t)(k0 + 32 + (tid >> 2)) * N + n0, 0, 1);
        if (w == 0) __builtin_amdgcn_s_wait_tensorcnt(0);
#else
        {   // fallback cooperative copies (row-major, no transpose)
            int lin = tid * 16;
            int r = lin >> 5, c = lin & 31;
            const __bf16* ap = A + (size_t)(m0 + r) * K + k0 + c;
            *(v8bf*)&lds_a[r][c]     = *(const v8bf*)ap;
            *(v8bf*)&lds_a[r][c + 8] = *(const v8bf*)(ap + 8);
            int kk = tid >> 2, nseg = (tid & 3) * 16;
            const __bf16* bp = B + (size_t)(k0 + kk) * N + n0 + nseg;
            *(v8bf*)&lds_b[kk][nseg]     = *(const v8bf*)bp;
            *(v8bf*)&lds_b[kk][nseg + 8] = *(const v8bf*)(bp + 8);
        }
#endif
        __syncthreads();

        // A fragment: row = 16w + ln; cols 16hl..16hl+15
        int row = w * 16 + ln;
        v16bf af = pack2(&lds_a[row][16 * hl], &lds_a[row][16 * hl + 8]);
#pragma unroll
        for (int nt = 0; nt < 4; nt++) {
            // B fragment: col = 16nt+ln; K rows 16hl..16hl+15 (strided gather)
            v16bf bfr;
#pragma unroll
            for (int j = 0; j < 16; j++) bfr[j] = lds_b[16 * hl + j][nt * 16 + ln];
            acc[nt] = wmma_bf16(af, bfr, acc[nt]);
        }
        __syncthreads();
    }

    // epilogue: C element (r,lane): row = r + 8hl, col = ln
#pragma unroll
    for (int nt = 0; nt < 4; nt++) {
#pragma unroll
        for (int r = 0; r < 8; r++) {
            int row = m0 + w * 16 + r + 8 * hl;
            int col = n0 + nt * 16 + ln;
            float v = acc[nt][r];
            if (OUT_BF16)
                ((__bf16*)Cout)[(size_t)row * N + col] = (__bf16)v;
            else
                ((float*)Cout)[(size_t)row * N + col] = v;
        }
    }
}

// ---------------------------------------------------------------------------
// RoPE applied in-place to Q and K (bf16, layout [B*S, NHEADS*HDIM]).
// ---------------------------------------------------------------------------
__global__ void rope_kernel(__bf16* __restrict__ Q, __bf16* __restrict__ K,
                            const int* __restrict__ pos_ids) {
    int idx = blockIdx.x * blockDim.x + threadIdx.x;  // BS_ROWS*16*32 threads
    int row = idx >> 9;
    int rem = idx & 511;
    int h   = rem >> 5;
    int d   = rem & 31;
    float pos = (float)pos_ids[row];
    float inv = __powf(10000.0f, -(float)d * (1.0f / 32.0f));
    float ang = pos * inv;
    float s, c;
    __sincosf(ang, &s, &c);
    size_t base = (size_t)row * DMODEL + h * HDIM;
    {
        float q0 = (float)Q[base + d], q1 = (float)Q[base + d + 32];
        Q[base + d]      = (__bf16)(q0 * c - q1 * s);
        Q[base + d + 32] = (__bf16)(q1 * c + q0 * s);
    }
    {
        float k0 = (float)K[base + d], k1 = (float)K[base + d + 32];
        K[base + d]      = (__bf16)(k0 * c - k1 * s);
        K[base + d + 32] = (__bf16)(k1 * c + k0 * s);
    }
}

// ---------------------------------------------------------------------------
// Flash attention: grid = (S/64, B*H), 128 threads = 4 waves.
// Per 32-key block: wave0 issues async TDM for the V tile (overlapped with
// the 4 score WMMAs + online softmax), P re-laid out through LDS, then 4
// P*V WMMAs. V fragments gathered transposed from the row-major TDM tile.
// ---------------------------------------------------------------------------
__global__ __launch_bounds__(128) void attn_kernel(
        const __bf16* __restrict__ Qp, const __bf16* __restrict__ Kp,
        const __bf16* __restrict__ Vp, const float* __restrict__ mask,
        __bf16* __restrict__ Out) {
    __shared__ __align__(16) __bf16 lds_p[4][16][32];   // per-wave P tile
    __shared__ __align__(16) __bf16 lds_v[32][64];      // V tile [key][dim]

    const int tid  = threadIdx.x;
    const int lane = tid & 31;
    const int w    = tid >> 5;
    const int hl   = lane >> 4;
    const int ln   = lane & 15;
    const int qt = blockIdx.x;          // 0..15
    const int bh = blockIdx.y;          // 0..63
    const int b  = bh >> 4, h = bh & 15;
    const int qrow0 = qt * 64 + w * 16;

    // Q fragments (held for whole kernel): head-dim chunks c=0,1
    const __bf16* qbase =
        Qp + (size_t)(b * S_LEN + qrow0 + ln) * DMODEL + h * HDIM;
    v16bf a0 = pack2(qbase + 16 * hl,      qbase + 16 * hl + 8);
    v16bf a1 = pack2(qbase + 32 + 16 * hl, qbase + 32 + 16 * hl + 8);

    float m_i[8], l_i[8];
    v8f acc[4];
#pragma unroll
    for (int r = 0; r < 8; r++) { m_i[r] = -1e30f; l_i[r] = 0.0f; }
#pragma unroll
    for (int i = 0; i < 4; i++) acc[i] = zero8();

#if HAVE_TDM
    const unsigned int lds_off_v = (unsigned int)(uintptr_t)&lds_v[0][0];
#endif

    for (int kb = 0; kb < S_LEN; kb += 32) {
#if HAVE_TDM
        if (w == 0)  // async V tile DMA: 32 keys x 64 dims, row stride DMODEL
            tdm_load_2d_bf16(lds_off_v,
                             Vp + (size_t)(b * S_LEN + kb) * DMODEL + h * HDIM,
                             (unsigned)DMODEL, (unsigned)(BATCH * S_LEN),
                             (unsigned)HDIM, 32u, (unsigned long long)DMODEL);
#else
        {
            int kk = tid >> 2, dseg = (tid & 3) * 16;
            const __bf16* vp =
                Vp + (size_t)(b * S_LEN + kb + kk) * DMODEL + h * HDIM + dseg;
            *(v8bf*)&lds_v[kk][dseg]     = *(const v8bf*)vp;
            *(v8bf*)&lds_v[kk][dseg + 8] = *(const v8bf*)(vp + 8);
        }
#endif

        // --- scores: two 16-key tiles, reduction over 64 head dims ---
        const __bf16* kbase0 =
            Kp + (size_t)(b * S_LEN + kb + ln) * DMODEL + h * HDIM + 16 * hl;
        const __bf16* kbase1 = kbase0 + (size_t)16 * DMODEL;
        v16bf bk00 = pack2(kbase0,      kbase0 + 8);
        v16bf bk01 = pack2(kbase0 + 32, kbase0 + 40);
        v16bf bk10 = pack2(kbase1,      kbase1 + 8);
        v16bf bk11 = pack2(kbase1 + 32, kbase1 + 40);
        v8f s0 = zero8(), s1 = zero8();
        s0 = wmma_bf16(a0, bk00, s0);
        s0 = wmma_bf16(a1, bk01, s0);
        s1 = wmma_bf16(a0, bk10, s1);
        s1 = wmma_bf16(a1, bk11, s1);

        // --- mask add + online softmax (row = r + 8*hl) ---
        const float* mrow =
            mask + ((size_t)bh * S_LEN + (qrow0 + 8 * hl)) * S_LEN + kb;
#pragma unroll
        for (int r = 0; r < 8; r++) {
            float x0 = s0[r] + mrow[(size_t)r * S_LEN + ln];
            float x1 = s1[r] + mrow[(size_t)r * S_LEN + 16 + ln];
            float mx = fmaxf(x0, x1);
#pragma unroll
            for (int m = 1; m < 16; m <<= 1)
                mx = fmaxf(mx, __shfl_xor(mx, m, 32));
            float mnew = fmaxf(m_i[r], mx);
            float corr = __expf(m_i[r] - mnew);
            float p0 = __expf(x0 - mnew);
            float p1 = __expf(x1 - mnew);
            float rs = p0 + p1;
#pragma unroll
            for (int m = 1; m < 16; m <<= 1) rs += __shfl_xor(rs, m, 32);
            l_i[r] = l_i[r] * corr + rs;
            m_i[r] = mnew;
#pragma unroll
            for (int nt = 0; nt < 4; nt++) acc[nt][r] *= corr;
            lds_p[w][r + 8 * hl][ln]      = (__bf16)p0;
            lds_p[w][r + 8 * hl][16 + ln] = (__bf16)p1;
        }
#if HAVE_TDM
        if (w == 0) __builtin_amdgcn_s_wait_tensorcnt(0);
#endif
        __syncthreads();

        // --- P fragment (A layout): row = ln, cols 16hl..16hl+15 ---
        v16bf pa = pack2(&lds_p[w][ln][16 * hl], &lds_p[w][ln][16 * hl + 8]);
#pragma unroll
        for (int nt = 0; nt < 4; nt++) {
            v16bf vb;  // col = 16nt+ln; key rows 16hl..16hl+15 (strided gather)
#pragma unroll
            for (int j = 0; j < 16; j++) vb[j] = lds_v[16 * hl + j][nt * 16 + ln];
            acc[nt] = wmma_bf16(pa, vb, acc[nt]);
        }
        __syncthreads();
    }

    // --- epilogue: normalize and store bf16 [B*S, NHEADS*HDIM] ---
#pragma unroll
    for (int nt = 0; nt < 4; nt++) {
#pragma unroll
        for (int r = 0; r < 8; r++) {
            float v = acc[nt][r] / l_i[r];
            int row = qrow0 + r + 8 * hl;
            Out[(size_t)(b * S_LEN + row) * DMODEL + h * HDIM + nt * 16 + ln] =
                (__bf16)v;
        }
    }
}

// ---------------------------------------------------------------------------
// Host launcher
// ---------------------------------------------------------------------------
extern "C" void kernel_launch(void* const* d_in, const int* in_sizes, int n_in,
                              void* d_out, int out_size, void* d_ws,
                              size_t ws_size, hipStream_t stream) {
    const float* hidden = (const float*)d_in[0];
    const float* mask   = (const float*)d_in[1];
    const int*   pos    = (const int*)d_in[2];
    const float* Wq     = (const float*)d_in[3];
    const float* Wk     = (const float*)d_in[4];
    const float* Wv     = (const float*)d_in[5];
    const float* Wo     = (const float*)d_in[6];
    float* out = (float*)d_out;

    char* ws = (char*)d_ws;
    size_t off = 0;
    const size_t actN = (size_t)BS_ROWS * DMODEL;
    const size_t wN   = (size_t)DMODEL * DMODEL;
    __bf16* Xb  = (__bf16*)(ws + off); off += actN * 2;
    __bf16* Wqb = (__bf16*)(ws + off); off += wN * 2;
    __bf16* Wkb = (__bf16*)(ws + off); off += wN * 2;
    __bf16* Wvb = (__bf16*)(ws + off); off += wN * 2;
    __bf16* Wob = (__bf16*)(ws + off); off += wN * 2;
    __bf16* Qp  = (__bf16*)(ws + off); off += actN * 2;
    __bf16* Kp  = (__bf16*)(ws + off); off += actN * 2;
    __bf16* Vp  = (__bf16*)(ws + off); off += actN * 2;
    __bf16* Ao  = (__bf16*)(ws + off); off += actN * 2;

    cvt_f32_bf16_kernel<<<(int)(actN / 256), 256, 0, stream>>>(hidden, Xb, (int)actN);
    cvt_f32_bf16_kernel<<<(int)(wN / 256), 256, 0, stream>>>(Wq, Wqb, (int)wN);
    cvt_f32_bf16_kernel<<<(int)(wN / 256), 256, 0, stream>>>(Wk, Wkb, (int)wN);
    cvt_f32_bf16_kernel<<<(int)(wN / 256), 256, 0, stream>>>(Wv, Wvb, (int)wN);
    cvt_f32_bf16_kernel<<<(int)(wN / 256), 256, 0, stream>>>(Wo, Wob, (int)wN);

    dim3 gg(BS_ROWS / 64, DMODEL / 64);
    gemm_wmma_kernel<true><<<gg, 128, 0, stream>>>(Xb, Wqb, Qp, BS_ROWS, DMODEL, DMODEL);
    gemm_wmma_kernel<true><<<gg, 128, 0, stream>>>(Xb, Wkb, Kp, BS_ROWS, DMODEL, DMODEL);
    gemm_wmma_kernel<true><<<gg, 128, 0, stream>>>(Xb, Wvb, Vp, BS_ROWS, DMODEL, DMODEL);

    int rope_threads = BS_ROWS * NHEADS * 32;
    rope_kernel<<<rope_threads / 256, 256, 0, stream>>>(Qp, Kp, pos);

    dim3 ga(S_LEN / 64, BATCH * NHEADS);
    attn_kernel<<<ga, 128, 0, stream>>>(Qp, Kp, Vp, mask, Ao);

    gemm_wmma_kernel<false><<<gg, 128, 0, stream>>>(Ao, Wob, out, BS_ROWS, DMODEL, DMODEL);
}
